// Attention_249108103377
// MI455X (gfx1250) — compile-verified, matrix operand order checked by reading, not run on hardware
//
#include <hip/hip_runtime.h>
#include <hip/hip_bf16.h>
#include <math.h>

typedef __bf16 bf16_t;
typedef __attribute__((ext_vector_type(16))) __bf16 v16bf;
typedef __attribute__((ext_vector_type(8)))  __bf16 v8bf;
typedef __attribute__((ext_vector_type(8)))  float  v8f;
typedef __attribute__((ext_vector_type(4)))  int    v4i;

typedef __attribute__((address_space(1))) v4i g_v4i;  // global int4
typedef __attribute__((address_space(3))) v4i l_v4i;  // LDS int4

#if defined(__has_builtin)
#if __has_builtin(__builtin_amdgcn_global_load_async_to_lds_b128)
#define HAVE_ASYNC_LDS 1
#endif
#endif

__device__ __forceinline__ void wait_asynccnt0() {
#if defined(HAVE_ASYNC_LDS)
#if __has_builtin(__builtin_amdgcn_s_wait_asynccnt)
  __builtin_amdgcn_s_wait_asynccnt(0);
#else
  asm volatile("s_wait_asynccnt 0x0" ::: "memory");
#endif
#endif
}

union V16U { v16bf v; v8bf h[2]; };

// Load a 16-wide (row/col striped) bf16 fragment whose 32 K-elements for this
// lane live contiguously at `base`. Matches CDNA5 16-bit A/B VGPR layout:
// elements 0..7 = base[hf*8 + 0..7], elements 8..15 = base[16 + hf*8 + 0..7].
__device__ __forceinline__ v16bf load_frag32(const bf16_t* base, int hf) {
  V16U f;
  f.h[0] = *(const v8bf*)(base + hf * 8);
  f.h[1] = *(const v8bf*)(base + 16 + hf * 8);
  return f.v;
}

__device__ __forceinline__ v8f wmma_bf16(v16bf a, v16bf b, v8f c) {
  return __builtin_amdgcn_wmma_f32_16x16x32_bf16(false, a, false, b, (short)0, c,
                                                 false, false);
}

// ln(10000)/32 for RoPE inv_freq: base^(-2i/64) = exp(-i * ln(10000)/32)
#define ROPE_LN_OVER_32 0.28782313663f

// ---------------------------------------------------------------------------
// Kernel 1: qkv = x @ w_qkv + b_qkv, fused RoPE on q/k, softmax scale folded
// into q. Output bf16 Q/K/V in [B, H, N, D] layout.
// Workgroup tile 256x64, wave tile 32x64 (8 WMMA per 32-K step).
// Grid: (8192/256, 3072/64) = (32, 48). Block: 256 (8 waves).
// ---------------------------------------------------------------------------
__global__ __launch_bounds__(256) void qkv_rope_kernel(
    const float* __restrict__ x, const float* __restrict__ w,
    const float* __restrict__ bias, bf16_t* __restrict__ Qs,
    bf16_t* __restrict__ Ks, bf16_t* __restrict__ Vs) {
  __shared__ __align__(16) bf16_t lds_x[256 * 32];   // [row][k]
  __shared__ __align__(16) bf16_t lds_wt[64 * 32];   // [col][k] (transposed)

  const int t = threadIdx.x;
  const int l = t & 31, wv = t >> 5;
  const int hf = l >> 4, nl = l & 15;
  const int rbase = blockIdx.x * 256;
  const int cb = blockIdx.y * 64;
  const int part = blockIdx.y >> 4;  // 0=q 1=k 2=v
  const int head = blockIdx.y & 15;

  v8f acc[2][4];
#pragma unroll
  for (int rb = 0; rb < 2; ++rb)
#pragma unroll
    for (int i = 0; i < 4; ++i) { v8f z = {}; acc[rb][i] = z; }

  for (int kb = 0; kb < 1024; kb += 32) {
    {  // stage x tile: one row per thread (fp32 -> bf16)
      const float* src = x + (size_t)(rbase + t) * 1024 + kb;
      if (kb + 32 < 1024) __builtin_prefetch(src + 32, 0, 1);
#pragma unroll
      for (int c = 0; c < 4; ++c) {
        v8bf p;
#pragma unroll
        for (int i = 0; i < 8; ++i) p[i] = (bf16_t)src[c * 8 + i];
        *(v8bf*)(lds_x + t * 32 + c * 8) = p;
      }
    }
    {  // stage w tile transposed (fp32 -> bf16)
      const int kl = t & 31, cs = (t >> 5) * 8;
      const float* src = w + (size_t)(kb + kl) * 3072 + cb + cs;
      if (kb + 32 < 1024) __builtin_prefetch(src + (size_t)32 * 3072, 0, 1);
#pragma unroll
      for (int i = 0; i < 8; ++i) lds_wt[(cs + i) * 32 + kl] = (bf16_t)src[i];
    }
    __syncthreads();

    v16bf bfr[4];
#pragma unroll
    for (int nb = 0; nb < 4; ++nb)
      bfr[nb] = load_frag32(lds_wt + (nb * 16 + nl) * 32, hf);
#pragma unroll
    for (int rb = 0; rb < 2; ++rb) {
      v16bf a = load_frag32(lds_x + (wv * 32 + rb * 16 + nl) * 32, hf);
#pragma unroll
      for (int nb = 0; nb < 4; ++nb)
        acc[rb][nb] = wmma_bf16(a, bfr[nb], acc[rb][nb]);
    }
    __syncthreads();
  }

  const float bias0 = bias[cb + nl];
  const float bias1 = bias[cb + 16 + nl];
  const float bias2 = bias[cb + 32 + nl];
  const float bias3 = bias[cb + 48 + nl];
  bf16_t* dst = (part == 0) ? Qs : ((part == 1) ? Ks : Vs);
  const float qscale = (part == 0) ? 0.125f : 1.0f;  // D^-0.5 folded into Q
  const float if0 = __expf(-(float)nl * ROPE_LN_OVER_32);
  const float if1 = __expf(-(float)(16 + nl) * ROPE_LN_OVER_32);

#pragma unroll
  for (int rb = 0; rb < 2; ++rb) {
#pragma unroll
    for (int r = 0; r < 8; ++r) {
      const int row = rbase + wv * 32 + rb * 16 + r + 8 * hf;
      const int b = row >> 11, pos = row & 2047;
      float c0 = acc[rb][0][r] + bias0;
      float c1 = acc[rb][1][r] + bias1;
      float c2 = acc[rb][2][r] + bias2;
      float c3 = acc[rb][3][r] + bias3;
      if (part < 2) {
        float s0, cs0, s1, cs1;
        __sincosf((float)pos * if0, &s0, &cs0);
        __sincosf((float)pos * if1, &s1, &cs1);
        const float o0 = c0 * cs0 - c2 * s0;  // d = nl     (rot = -x[d+32])
        const float o1 = c1 * cs1 - c3 * s1;  // d = 16+nl
        const float o2 = c2 * cs0 + c0 * s0;  // d = 32+nl  (rot =  x[d-32])
        const float o3 = c3 * cs1 + c1 * s1;  // d = 48+nl
        c0 = o0; c1 = o1; c2 = o2; c3 = o3;
      }
      const size_t base = (((size_t)b * 16 + head) * 2048 + pos) * 64;
      dst[base + nl]      = (bf16_t)(c0 * qscale);
      dst[base + 16 + nl] = (bf16_t)(c1 * qscale);
      dst[base + 32 + nl] = (bf16_t)(c2 * qscale);
      dst[base + 48 + nl] = (bf16_t)(c3 * qscale);
    }
  }
}

// ---------------------------------------------------------------------------
// Kernel 2: flash attention per (b,h). Each wave owns 16 query rows; loop over
// 64-key blocks with online softmax (16 WMMA per block). K tile staged with
// CDNA5 async global->LDS copies when available (ASYNCcnt-tracked).
// Output bf16 [B, N, C] for the projection.
// Grid: (2048/128, B*H) = (16, 64). Block: 256 (8 waves).
// ---------------------------------------------------------------------------
__global__ __launch_bounds__(256) void attn_kernel(
    const bf16_t* __restrict__ Qs, const bf16_t* __restrict__ Ks,
    const bf16_t* __restrict__ Vs, bf16_t* __restrict__ AO) {
  __shared__ __align__(16) bf16_t lds_k[64 * 64];     // [key][d]
  __shared__ __align__(16) bf16_t lds_vt[64 * 64];    // [d][key] transposed
  __shared__ __align__(16) bf16_t lds_p[8 * 16 * 64]; // per-wave P tile

  const int t = threadIdx.x, l = t & 31, wv = t >> 5;
  const int hf = l >> 4, nl = l & 15;
  const int bh = blockIdx.y;
  const int b = bh >> 4, head = bh & 15;
  const int qbase = blockIdx.x * 128 + wv * 16;
  const size_t hb = (size_t)bh * 2048;

  v16bf aq[2];
#pragma unroll
  for (int kc = 0; kc < 2; ++kc)
    aq[kc] = load_frag32(Qs + (hb + qbase + nl) * 64 + kc * 32, hf);

  v8f o[4];
#pragma unroll
  for (int i = 0; i < 4; ++i) { v8f z = {}; o[i] = z; }
  float mrun[8], lrun[8];
#pragma unroll
  for (int r = 0; r < 8; ++r) { mrun[r] = -__builtin_inff(); lrun[r] = 0.0f; }

  for (int kb = 0; kb < 2048; kb += 64) {
    {  // stage K (rows, async when available) and V (transposed): 64x64
      const int kr = t >> 2, sg = (t & 3) * 16;
      const bf16_t* ksrc = Ks + (hb + kb + kr) * 64 + sg;
      bf16_t* kdst = lds_k + kr * 64 + sg;
#if defined(HAVE_ASYNC_LDS)
      __builtin_amdgcn_global_load_async_to_lds_b128(
          (g_v4i*)ksrc, (l_v4i*)kdst, 0, 0);
      __builtin_amdgcn_global_load_async_to_lds_b128(
          (g_v4i*)(ksrc + 8), (l_v4i*)(kdst + 8), 0, 0);
#else
      *(v8bf*)kdst = *(const v8bf*)ksrc;
      *(v8bf*)(kdst + 8) = *(const v8bf*)(ksrc + 8);
#endif
      const bf16_t* vsrc = Vs + (hb + kb + kr) * 64 + sg;
      v8bf v0 = *(const v8bf*)vsrc;
      v8bf v1 = *(const v8bf*)(vsrc + 8);
#pragma unroll
      for (int i = 0; i < 8; ++i) {
        lds_vt[(sg + i) * 64 + kr] = v0[i];
        lds_vt[(sg + 8 + i) * 64 + kr] = v1[i];
      }
      wait_asynccnt0();  // this wave's async K copies landed in LDS
    }
    __syncthreads();

    // S = Q K^T  (16 rows x 64 keys per wave)
    v8f s[4];
#pragma unroll
    for (int i = 0; i < 4; ++i) { v8f z = {}; s[i] = z; }
#pragma unroll
    for (int kc = 0; kc < 2; ++kc) {
#pragma unroll
      for (int nb = 0; nb < 4; ++nb) {
        v16bf bk = load_frag32(lds_k + (nb * 16 + nl) * 64 + kc * 32, hf);
        s[nb] = wmma_bf16(aq[kc], bk, s[nb]);
      }
    }

    // online softmax; write P (bf16) to per-wave LDS for C->A re-layout
#pragma unroll
    for (int r = 0; r < 8; ++r) {
      float mb = fmaxf(fmaxf(s[0][r], s[1][r]), fmaxf(s[2][r], s[3][r]));
#pragma unroll
      for (int off = 8; off > 0; off >>= 1)
        mb = fmaxf(mb, __shfl_xor(mb, off, 16));
      const float mnew = fmaxf(mrun[r], mb);
      const float alpha = __expf(mrun[r] - mnew);
      const float p0 = __expf(s[0][r] - mnew);
      const float p1 = __expf(s[1][r] - mnew);
      const float p2 = __expf(s[2][r] - mnew);
      const float p3 = __expf(s[3][r] - mnew);
      float rs = (p0 + p1) + (p2 + p3);
#pragma unroll
      for (int off = 8; off > 0; off >>= 1) rs += __shfl_xor(rs, off, 16);
      lrun[r] = lrun[r] * alpha + rs;
      mrun[r] = mnew;
#pragma unroll
      for (int db = 0; db < 4; ++db) o[db][r] = o[db][r] * alpha;
      const int m = r + 8 * hf;
      bf16_t* pr = lds_p + wv * 1024 + m * 64 + nl;
      pr[0]  = (bf16_t)p0;
      pr[16] = (bf16_t)p1;
      pr[32] = (bf16_t)p2;
      pr[48] = (bf16_t)p3;
    }

    // O += P V
#pragma unroll
    for (int kc = 0; kc < 2; ++kc) {
      v16bf ap = load_frag32(lds_p + wv * 1024 + nl * 64 + kc * 32, hf);
#pragma unroll
      for (int db = 0; db < 4; ++db) {
        v16bf bv = load_frag32(lds_vt + (db * 16 + nl) * 64 + kc * 32, hf);
        o[db] = wmma_bf16(ap, bv, o[db]);
      }
    }
    __syncthreads();
  }

#pragma unroll
  for (int r = 0; r < 8; ++r) {
    const float inv = 1.0f / lrun[r];
    const int pos = qbase + r + 8 * hf;
    const size_t obase = ((size_t)b * 2048 + pos) * 1024 + head * 64;
#pragma unroll
    for (int db = 0; db < 4; ++db)
      AO[obase + db * 16 + nl] = (bf16_t)(o[db][r] * inv);
  }
}

// ---------------------------------------------------------------------------
// Kernel 3: out = AO @ w_proj + b_proj  (fp32 output).
// Workgroup tile 256x64, wave tile 32x64.
// Grid: (8192/256, 1024/64) = (32, 16). Block: 256 (8 waves).
// ---------------------------------------------------------------------------
__global__ __launch_bounds__(256) void proj_kernel(
    const bf16_t* __restrict__ A, const float* __restrict__ w,
    const float* __restrict__ bias, float* __restrict__ out) {
  __shared__ __align__(16) bf16_t lds_a[256 * 32];
  __shared__ __align__(16) bf16_t lds_wt[64 * 32];

  const int t = threadIdx.x, l = t & 31, wv = t >> 5;
  const int hf = l >> 4, nl = l & 15;
  const int rbase = blockIdx.x * 256;
  const int cb = blockIdx.y * 64;

  v8f acc[2][4];
#pragma unroll
  for (int rb = 0; rb < 2; ++rb)
#pragma unroll
    for (int i = 0; i < 4; ++i) { v8f z = {}; acc[rb][i] = z; }

  for (int kb = 0; kb < 1024; kb += 32) {
    {  // stage A tile: one row per thread (already bf16)
      const bf16_t* src = A + (size_t)(rbase + t) * 1024 + kb;
      if (kb + 32 < 1024) __builtin_prefetch(src + 32, 0, 1);
#pragma unroll
      for (int c = 0; c < 4; ++c)
        *(v8bf*)(lds_a + t * 32 + c * 8) = *(const v8bf*)(src + c * 8);
    }
    {  // stage w tile transposed (fp32 -> bf16)
      const int kl = t & 31, cs = (t >> 5) * 8;
      const float* src = w + (size_t)(kb + kl) * 1024 + cb + cs;
      if (kb + 32 < 1024) __builtin_prefetch(src + (size_t)32 * 1024, 0, 1);
#pragma unroll
      for (int i = 0; i < 8; ++i) lds_wt[(cs + i) * 32 + kl] = (bf16_t)src[i];
    }
    __syncthreads();

    v16bf bfr[4];
#pragma unroll
    for (int nb = 0; nb < 4; ++nb)
      bfr[nb] = load_frag32(lds_wt + (nb * 16 + nl) * 32, hf);
#pragma unroll
    for (int rb = 0; rb < 2; ++rb) {
      v16bf a = load_frag32(lds_a + (wv * 32 + rb * 16 + nl) * 32, hf);
#pragma unroll
      for (int nb = 0; nb < 4; ++nb)
        acc[rb][nb] = wmma_bf16(a, bfr[nb], acc[rb][nb]);
    }
    __syncthreads();
  }

#pragma unroll
  for (int nb = 0; nb < 4; ++nb) {
    const float bv = bias[cb + nb * 16 + nl];
#pragma unroll
    for (int rb = 0; rb < 2; ++rb) {
#pragma unroll
      for (int r = 0; r < 8; ++r) {
        const int row = rbase + wv * 32 + rb * 16 + r + 8 * hf;
        out[(size_t)row * 1024 + cb + nb * 16 + nl] = acc[rb][nb][r] + bv;
      }
    }
  }
}

extern "C" void kernel_launch(void* const* d_in, const int* in_sizes, int n_in,
                              void* d_out, int out_size, void* d_ws,
                              size_t ws_size, hipStream_t stream) {
  const float* x = (const float*)d_in[0];
  const float* w_qkv = (const float*)d_in[1];
  const float* b_qkv = (const float*)d_in[2];
  const float* w_proj = (const float*)d_in[3];
  const float* b_proj = (const float*)d_in[4];
  float* out = (float*)d_out;

  const size_t HEAD_ELEMS = (size_t)4 * 16 * 2048 * 64;  // 8388608 per tensor
  bf16_t* Qs = (bf16_t*)d_ws;
  bf16_t* Ks = Qs + HEAD_ELEMS;
  bf16_t* Vs = Ks + HEAD_ELEMS;
  bf16_t* AO = Vs + HEAD_ELEMS;

  qkv_rope_kernel<<<dim3(32, 48), 256, 0, stream>>>(x, w_qkv, b_qkv, Qs, Ks, Vs);
  attn_kernel<<<dim3(16, 64), 256, 0, stream>>>(Qs, Ks, Vs, AO);
  proj_kernel<<<dim3(32, 16), 256, 0, stream>>>(AO, w_proj, b_proj, out);
}
